// Attention_48189533061798
// MI455X (gfx1250) — compile-verified, matrix operand order checked by reading, not run on hardware
//
#include <hip/hip_runtime.h>
#include <hip/hip_bf16.h>
#include <cstdint>
#include <cstddef>

// Problem constants (match reference)
#define B_   2
#define S_   2048
#define DIM_ 4096
#define NH_  32
#define NKV_ 8
#define HD_  128
#define REP_ (NH_ / NKV_)

typedef __attribute__((ext_vector_type(16))) __bf16       v16bf;
typedef __attribute__((ext_vector_type(8)))  float        v8f;
typedef __attribute__((ext_vector_type(8)))  unsigned int v8u;
typedef __attribute__((ext_vector_type(4)))  unsigned int u32x4;
typedef __attribute__((ext_vector_type(8)))  int          i32x8;
typedef __attribute__((ext_vector_type(4)))  int          i32x4;

#if defined(__HIP_DEVICE_COMPILE__) && __has_builtin(__builtin_amdgcn_tensor_load_to_lds)
#define HAVE_TDM 1
#endif

// ---------------------------------------------------------------- helpers
__device__ __forceinline__ unsigned short f2bf(float f) {
  unsigned int x = __builtin_bit_cast(unsigned int, f);
  x += 0x7FFFu + ((x >> 16) & 1u);           // round-to-nearest-even
  return (unsigned short)(x >> 16);
}
__device__ __forceinline__ float bf2f(unsigned short u) {
  unsigned int x = ((unsigned int)u) << 16;
  return __builtin_bit_cast(float, x);
}

__device__ __forceinline__ v8f wmma_bf16(v16bf a, v16bf b, v8f c) {
  return __builtin_amdgcn_wmma_f32_16x16x32_bf16(
      /*neg_a=*/false, a, /*neg_b=*/false, b,
      /*c_mod=*/(short)0, c, /*reuse_a=*/false, /*reuse_b=*/false);
}

// Load one 16x32 bf16 WMMA operand (A layout; B uses the same pattern with
// "row" meaning the N column). Row-major source with leading dim `ld`
// (elements). Lane L<16: row=row0+L, k={kk..kk+7, kk+16..kk+23};
// Lane L>=16: row=row0+L-16, k={kk+8..kk+15, kk+24..kk+31}.
__device__ __forceinline__ v16bf load_op16(const unsigned short* base, int ld,
                                           int row0, int kk, int lane) {
  const int half = lane >> 4;
  const unsigned short* p =
      base + (size_t)(row0 + (lane & 15)) * (size_t)ld + kk + half * 8;
  const uint32_t* p32 = (const uint32_t*)p;
  v8u u;
  u[0] = p32[0]; u[1] = p32[1]; u[2] = p32[2]; u[3] = p32[3];
  u[4] = p32[8]; u[5] = p32[9]; u[6] = p32[10]; u[7] = p32[11];
  return __builtin_bit_cast(v16bf, u);
}

#ifdef HAVE_TDM
// Tensor Data Mover: DMA a 2-D bf16 tile [tile_d1 rows x tile_d0 elems] whose
// rows are `stride0` elements apart in global memory into contiguous
// row-major LDS at byte offset lds_off. D# packing per CDNA5 ISA 8.3/8.4.
__device__ __forceinline__ void tdm_load_2d_bf16(uint32_t lds_off,
                                                 const void* gptr,
                                                 uint32_t tile_d0,
                                                 uint32_t tile_d1,
                                                 uint64_t stride0) {
  const uint64_t ga = (uint64_t)(uintptr_t)gptr;
  u32x4 g0;
  g0[0] = 1u;                                           // count=1, user D#
  g0[1] = lds_off;                                      // lds byte address
  g0[2] = (uint32_t)(ga & 0xFFFFFFFFu);                 // global_addr[31:0]
  g0[3] = (uint32_t)((ga >> 32) & 0x1FFFFFFu) | (2u << 30); // [56:32] | type=2
  i32x8 g1;
  g1[0] = (int)(1u << 16);                              // data_size=1 (2 bytes)
  g1[1] = (int)((tile_d0 & 0xFFFFu) << 16);             // tensor_dim0[15:0]
  g1[2] = (int)((tile_d0 >> 16) | ((tile_d1 & 0xFFFFu) << 16)); // dim0 hi | dim1 lo
  g1[3] = (int)((tile_d1 >> 16) | (tile_d0 << 16));     // dim1 hi | tile_dim0
  g1[4] = (int)tile_d1;                                 // tile_dim1 (tile_dim2=0)
  g1[5] = (int)(uint32_t)(stride0 & 0xFFFFFFFFu);       // tensor_dim0_stride lo
  g1[6] = (int)(uint32_t)((stride0 >> 32) & 0xFFFFu);   // stride hi (dim1_stride=0)
  g1[7] = 0;
  i32x4 z4 = {0, 0, 0, 0};
#if __clang_major__ >= 23
  i32x8 z8 = {0, 0, 0, 0, 0, 0, 0, 0};
  __builtin_amdgcn_tensor_load_to_lds(g0, g1, z4, z4, z8, 0);
#else
  __builtin_amdgcn_tensor_load_to_lds(g0, g1, z4, z4, 0);
#endif
}
#endif

// ---------------------------------------------------------------- fp32 -> bf16
__global__ void cast_f32_bf16(const float* __restrict__ in,
                              unsigned short* __restrict__ out, int n) {
  int i = blockIdx.x * blockDim.x + threadIdx.x;
  if (i < n) out[i] = f2bf(in[i]);
}

// ---------------------------------------------------------------- GEMM
// C[M,N] = A[M,K] (bf16 row-major) x W[N,K]^T (bf16 row-major).
// 128 threads = 4 waves in a 2x2 grid; block tile 64(M) x 128(N);
// wave tile 32 x 64 => 8 f32 16x16 accumulators, 8 WMMA per 32-k step.
// Operands stream from the 192 MB L2 (entire bf16 working set is resident).
template <bool BF16OUT>
__global__ void __launch_bounds__(128, 2)
gemm_wmma(const unsigned short* __restrict__ A,
          const unsigned short* __restrict__ W,
          void* __restrict__ C, int M, int N, int K) {
  const int lane  = threadIdx.x & 31;
  const int wave  = threadIdx.x >> 5;
  const int waveM = wave >> 1;     // 0..1
  const int waveN = wave & 1;      // 0..1
  const int m0 = blockIdx.y * 64  + waveM * 32;
  const int n0 = blockIdx.x * 128 + waveN * 64;

  v8f acc[2][4] = {};
#pragma unroll 2
  for (int kk = 0; kk < K; kk += 32) {
    v16bf a0 = load_op16(A, K, m0,      kk, lane);
    v16bf a1 = load_op16(A, K, m0 + 16, kk, lane);
    v16bf b0 = load_op16(W, K, n0,      kk, lane);
    v16bf b1 = load_op16(W, K, n0 + 16, kk, lane);
    v16bf b2 = load_op16(W, K, n0 + 32, kk, lane);
    v16bf b3 = load_op16(W, K, n0 + 48, kk, lane);
    acc[0][0] = wmma_bf16(a0, b0, acc[0][0]);
    acc[0][1] = wmma_bf16(a0, b1, acc[0][1]);
    acc[0][2] = wmma_bf16(a0, b2, acc[0][2]);
    acc[0][3] = wmma_bf16(a0, b3, acc[0][3]);
    acc[1][0] = wmma_bf16(a1, b0, acc[1][0]);
    acc[1][1] = wmma_bf16(a1, b1, acc[1][1]);
    acc[1][2] = wmma_bf16(a1, b2, acc[1][2]);
    acc[1][3] = wmma_bf16(a1, b3, acc[1][3]);
  }

  const int col  = lane & 15;
  const int half = lane >> 4;
#pragma unroll
  for (int i = 0; i < 2; ++i)
#pragma unroll
    for (int j = 0; j < 4; ++j)
#pragma unroll
      for (int r = 0; r < 8; ++r) {
        const size_t mm = (size_t)(m0 + i * 16 + r + half * 8);
        const size_t nn = (size_t)(n0 + j * 16 + col);
        const float v = acc[i][j][r];
        if (BF16OUT)
          ((unsigned short*)C)[mm * (size_t)N + nn] = f2bf(v);
        else
          ((float*)C)[mm * (size_t)N + nn] = v;
      }
}

// ---------------------------------------------------------------- RoPE
// In-place interleaved rotary on bf16 tensor [B, S, nheads, HD].
__global__ void rope_bf16(unsigned short* __restrict__ t,
                          const float* __restrict__ fcos,
                          const float* __restrict__ fsin, int nheads, int total) {
  int idx = blockIdx.x * blockDim.x + threadIdx.x;   // one (b,s,h,pair)
  if (idx >= total) return;
  const int HP = HD_ / 2;
  int i = idx % HP;
  int h = (idx / HP) % nheads;
  int s = (idx / (HP * nheads)) % S_;
  int b = idx / (HP * nheads * S_);
  float c  = fcos[s * HP + i];
  float sn = fsin[s * HP + i];
  size_t base = (((size_t)b * S_ + s) * nheads + h) * HD_ + 2 * i;
  uint32_t* p = (uint32_t*)(t + base);
  uint32_t w = *p;
  float xr = bf2f((unsigned short)(w & 0xFFFFu));
  float xi = bf2f((unsigned short)(w >> 16));
  float orr = xr * c - xi * sn;
  float oii = xr * sn + xi * c;
  *p = (uint32_t)f2bf(orr) | ((uint32_t)f2bf(oii) << 16);
}

// ---------------------------------------------------------------- flash attention
// Grid: (S/64, NH, B). Block: 128 threads = 4 waves; wave w owns q rows
// [q0+16w, q0+16w+15]. Q kept in registers (4 A-operands). K tile staged by
// the Tensor Data Mover (double-buffered, issued one tile ahead by wave 0);
// V tile staged transposed by the VALU path. Online softmax uses 16-lane
// xor-shuffle reductions that match the WMMA C layout. Operand loads are
// batched ahead of the WMMA chains so DS latency overlaps compute.
__global__ void __launch_bounds__(128, 1)
flash_attn_wmma(const unsigned short* __restrict__ Qb,
                const unsigned short* __restrict__ Kb,
                const unsigned short* __restrict__ Vb,
                unsigned short* __restrict__ Ob) {
  __shared__ __align__(16) unsigned short ktile[2][32 * 128];  // [buf][key][d]
  __shared__ __align__(16) unsigned short vtile[128 * 32];     // [d][key]
  __shared__ __align__(16) unsigned short ptile[4 * 16 * 32];  // per-wave P

  const int lane = threadIdx.x & 31;
  const int wave = threadIdx.x >> 5;
  const int b  = blockIdx.z;
  const int h  = blockIdx.y;
  const int hk = h / REP_;                 // GQA group
  const int q0 = blockIdx.x * 64;
  const int qw = q0 + wave * 16;

  const unsigned short* Qp = Qb + ((size_t)b * S_ * NH_  + h ) * HD_;
  const unsigned short* Kp = Kb + ((size_t)b * S_ * NKV_ + hk) * HD_;
  const unsigned short* Vp = Vb + ((size_t)b * S_ * NKV_ + hk) * HD_;

  // Q operands: d-chunks of 32, row stride NH_*HD_
  v16bf qa[4];
#pragma unroll
  for (int c = 0; c < 4; ++c)
    qa[c] = load_op16(Qp, NH_ * HD_, qw, c * 32, lane);

  v8f   o[8] = {};
  float m_i[8], l_i[8];
#pragma unroll
  for (int r = 0; r < 8; ++r) { m_i[r] = -1.0e30f; l_i[r] = 0.0f; }

  const float scale = 0.08838834764831845f;   // 1/sqrt(128)
  const int   col   = lane & 15;
  const int   half  = lane >> 4;
  const int jtiles  = q0 / 32 + 2;            // keys [0, q0+63]

#ifdef HAVE_TDM
  if (wave == 0)   // prologue: DMA K tile 0 into buffer 0
    tdm_load_2d_bf16((uint32_t)(uintptr_t)&ktile[0][0], Kp, 128, 32,
                     (uint64_t)(NKV_ * HD_));
#endif

  for (int j = 0; j < jtiles; ++j) {
    const int kbase = j * 32;
    const int jb    = j & 1;
    __syncthreads();                          // previous tile fully consumed
    {
      // V staging (transposed): thread t handles key=(t>>2), d=[(t&3)*32..+31]
      const int t   = threadIdx.x;
      const int key = t >> 2;
      const int db  = (t & 3) * 32;
      const unsigned short* vs =
          Vp + (size_t)(kbase + key) * (NKV_ * HD_) + db;
#pragma unroll
      for (int i = 0; i < 32; ++i) vtile[(db + i) * 32 + key] = vs[i];
#ifndef HAVE_TDM
      const uint32_t* src =
          (const uint32_t*)(Kp + (size_t)(kbase + key) * (NKV_ * HD_) + db);
      uint32_t* dst = (uint32_t*)(&ktile[jb][0] + key * 128 + db);
#pragma unroll
      for (int i = 0; i < 16; ++i) dst[i] = src[i];
#endif
    }
#ifdef HAVE_TDM
    if (wave == 0) {
      if (j + 1 < jtiles) {   // prefetch next tile into the other buffer
        tdm_load_2d_bf16((uint32_t)(uintptr_t)&ktile[(j + 1) & 1][0],
                         Kp + (size_t)(kbase + 32) * (NKV_ * HD_), 128, 32,
                         (uint64_t)(NKV_ * HD_));
        __builtin_amdgcn_s_wait_tensorcnt(1);  // in-order => tile j landed
      } else {
        __builtin_amdgcn_s_wait_tensorcnt(0);
      }
    }
#endif
    __syncthreads();

    if (kbase <= qw + 15) {                   // wave-uniform causal skip
      const unsigned short* kt = &ktile[jb][0];

      // Batch all 8 K-operands, then run the 8 score WMMAs back-to-back so
      // the DS clause overlaps the matrix pipe instead of lock-stepping.
      v16bf kb[8];
#pragma unroll
      for (int c = 0; c < 4; ++c) {
        kb[2 * c]     = load_op16(kt, 128, 0,  c * 32, lane);
        kb[2 * c + 1] = load_op16(kt, 128, 16, c * 32, lane);
      }
      v8f s0 = {}, s1 = {};
#pragma unroll
      for (int c = 0; c < 4; ++c) {
        s0 = wmma_bf16(qa[c], kb[2 * c],     s0);
        s1 = wmma_bf16(qa[c], kb[2 * c + 1], s1);
      }

      // Issue the 8 V-operand loads now: independent of the softmax VALU
      // work below, so their LDS latency is fully hidden.
      v16bf vb[8];
#pragma unroll
      for (int n = 0; n < 8; ++n)
        vb[n] = load_op16(vtile, 32, n * 16, 0, lane);

      unsigned short* Pw = ptile + wave * (16 * 32);
      float alpha[8];
#pragma unroll
      for (int r = 0; r < 8; ++r) {
        const int qrow = qw + r + half * 8;
        const int k0 = kbase + col;
        const int k1 = kbase + 16 + col;
        float v0 = (k0 <= qrow) ? s0[r] * scale : -1.0e30f;
        float v1 = (k1 <= qrow) ? s1[r] * scale : -1.0e30f;
        float mx = fmaxf(v0, v1);
#pragma unroll
        for (int msk = 1; msk < 16; msk <<= 1)
          mx = fmaxf(mx, __shfl_xor(mx, msk, 32));
        const float mnew = fmaxf(m_i[r], mx);
        const float a = __expf(m_i[r] - mnew);
        const float p0 = __expf(v0 - mnew);
        const float p1 = __expf(v1 - mnew);
        float rs = p0 + p1;
#pragma unroll
        for (int msk = 1; msk < 16; msk <<= 1)
          rs += __shfl_xor(rs, msk, 32);
        l_i[r]  = l_i[r] * a + rs;
        m_i[r]  = mnew;
        alpha[r] = a;
        const int prow = r + half * 8;
        Pw[prow * 32 + col]      = f2bf(p0);
        Pw[prow * 32 + col + 16] = f2bf(p1);
      }
#pragma unroll
      for (int n = 0; n < 8; ++n)
#pragma unroll
        for (int r = 0; r < 8; ++r) o[n][r] *= alpha[r];

      asm volatile("s_wait_dscnt 0" ::: "memory");  // P writes visible to reads

      // O += P(16x32) @ V(32x128): A = P, B = Vt d-chunks (already loaded)
      v16bf pa = load_op16(Pw, 32, 0, 0, lane);
#pragma unroll
      for (int n = 0; n < 8; ++n)
        o[n] = wmma_bf16(pa, vb[n], o[n]);
    }
  }

  // epilogue: normalize and store bf16 [b][q][h][d]
#pragma unroll
  for (int n = 0; n < 8; ++n)
#pragma unroll
    for (int r = 0; r < 8; ++r) {
      const int row = qw + r + half * 8;
      const int d   = n * 16 + col;
      const float v = o[n][r] / l_i[r];
      Ob[((size_t)b * S_ + row) * (NH_ * HD_) + h * HD_ + d] = f2bf(v);
    }
}

// ---------------------------------------------------------------- launch
extern "C" void kernel_launch(void* const* d_in, const int* in_sizes, int n_in,
                              void* d_out, int out_size, void* d_ws, size_t ws_size,
                              hipStream_t stream) {
  const float* x    = (const float*)d_in[0];
  const float* wq   = (const float*)d_in[1];
  const float* wk   = (const float*)d_in[2];
  const float* wv   = (const float*)d_in[3];
  const float* wo   = (const float*)d_in[4];
  const float* fcos = (const float*)d_in[5];
  const float* fsin = (const float*)d_in[6];
  (void)in_sizes; (void)n_in; (void)out_size; (void)ws_size;

  // bf16 workspace layout; xA region is reused: x early, attn-out late;
  // Wbig is reused: wq early, wo late. Total = 128 MB.
  char* ws = (char*)d_ws;
  const size_t nX  = (size_t)B_ * S_ * DIM_;
  const size_t nWq = (size_t)NH_ * HD_ * DIM_;
  const size_t nWk = (size_t)NKV_ * HD_ * DIM_;
  const size_t nQ  = (size_t)B_ * S_ * NH_ * HD_;
  const size_t nK  = (size_t)B_ * S_ * NKV_ * HD_;

  unsigned short* xA   = (unsigned short*)ws;                 // x, later attn out
  unsigned short* Wbig = xA   + nX;                           // wq, later wo
  unsigned short* wkb  = Wbig + nWq;
  unsigned short* wvb  = wkb  + nWk;
  unsigned short* Qb   = wvb  + nWk;
  unsigned short* Kb   = Qb   + nQ;
  unsigned short* Vb   = Kb   + nK;

  const int TPB = 256;
  // 1) casts
  cast_f32_bf16<<<dim3((unsigned)((nX  + TPB - 1) / TPB)), TPB, 0, stream>>>(x,  xA,   (int)nX);
  cast_f32_bf16<<<dim3((unsigned)((nWq + TPB - 1) / TPB)), TPB, 0, stream>>>(wq, Wbig, (int)nWq);
  cast_f32_bf16<<<dim3((unsigned)((nWk + TPB - 1) / TPB)), TPB, 0, stream>>>(wk, wkb,  (int)nWk);
  cast_f32_bf16<<<dim3((unsigned)((nWk + TPB - 1) / TPB)), TPB, 0, stream>>>(wv, wvb,  (int)nWk);

  // 2) QKV projections (bf16 out)
  const int M = B_ * S_;
  gemm_wmma<true><<<dim3(NH_ * HD_ / 128, M / 64),  128, 0, stream>>>(xA, Wbig, Qb, M, NH_ * HD_,  DIM_);
  gemm_wmma<true><<<dim3(NKV_ * HD_ / 128, M / 64), 128, 0, stream>>>(xA, wkb,  Kb, M, NKV_ * HD_, DIM_);
  gemm_wmma<true><<<dim3(NKV_ * HD_ / 128, M / 64), 128, 0, stream>>>(xA, wvb,  Vb, M, NKV_ * HD_, DIM_);

  // 3) RoPE on Q and K (in place, bf16)
  const int qPairs = B_ * S_ * NH_  * (HD_ / 2);
  const int kPairs = B_ * S_ * NKV_ * (HD_ / 2);
  rope_bf16<<<dim3((qPairs + TPB - 1) / TPB), TPB, 0, stream>>>(Qb, fcos, fsin, NH_,  qPairs);
  rope_bf16<<<dim3((kPairs + TPB - 1) / TPB), TPB, 0, stream>>>(Kb, fcos, fsin, NKV_, kPairs);

  // 4) wo cast into Wbig (wq no longer needed)
  cast_f32_bf16<<<dim3((unsigned)((nWq + TPB - 1) / TPB)), TPB, 0, stream>>>(wo, Wbig, (int)nWq);

  // 5) flash attention -> attn out (bf16) into xA region (x no longer needed)
  flash_attn_wmma<<<dim3(S_ / 64, NH_, B_), 128, 0, stream>>>(Qb, Kb, Vb, xA);

  // 6) output projection, fp32 result straight to d_out
  gemm_wmma<false><<<dim3(DIM_ / 128, M / 64), 128, 0, stream>>>(xA, Wbig, d_out, M, DIM_, NH_ * HD_);
}
